// MIL_SB_5901285064952
// MI455X (gfx1250) — matrix operand order repaired and checked by reading, not compiled, not run on hardware
//
#include <hip/hip_runtime.h>
#include <cmath>

// ---------------- problem constants ----------------
#define N_PATCH 50000
#define D_IN    2048
#define H1_     512
#define H2_     256
#define K_SEL   20

#define M_ROWS  64   // patch rows per workgroup
#define MT      4    // M subtiles of 16
#define NSTAGE  (D_IN / 128)

// ---------------- d_out layout (float offsets, reference return order) ----
#define OUT_LOGITS 0
#define OUT_YPROB  2
#define OUT_YHAT   4
#define OUT_ARAW   5
#define OUT_ILOSS  50005
#define OUT_PREDS0 50006
#define OUT_PREDS1 50007
#define OUT_HF     50067

// ---------------- workspace layout (byte offsets) ----------------
#define WS_WFCT  0u         // bf16 [H1][D]   2048*512*2 = 2,097,152
#define WS_WVT   2097152u   // bf16 [H2][H1]  256*512*2  =   262,144
#define WS_WUT   2359296u   // bf16 [H2][H1]
#define WS_STATS 2621440u   // f32 amax, f32 Z
#define WS_M     2621504u   // f32 [512]
#define WS_IDX   2623552u   // int  [64]   top20 | bot20 | mid20
#define WS_FLAG  2623808u   // u8   [50048]

typedef __attribute__((ext_vector_type(16))) __bf16 v16bf;
typedef __attribute__((ext_vector_type(8)))  float  v8f;

union Frag { uint4 q[2]; v16bf v; };

// ---------------- kernel 1: fp32 [R][C] -> bf16 transposed [C][R] ----------
__global__ __launch_bounds__(256) void transpose_cvt(
    const float* __restrict__ src, __bf16* __restrict__ dst, int R, int C) {
  int i = blockIdx.x * 256 + threadIdx.x;
  if (i < R * C) {
    int r = i / C, c = i % C;
    dst[(size_t)c * R + r] = (__bf16)src[i];
  }
}

// ---------------- kernel 2: fused feats + gated attention ------------------
// grid = ceil(N_PATCH/64) workgroups of 256 threads (8 waves): 64 rows each.
__global__ __launch_bounds__(256) void fused_attn(
    const float* __restrict__ h,
    const float* __restrict__ b_fc, const float* __restrict__ b_v,
    const float* __restrict__ b_u,  const float* __restrict__ w_a,
    const float* __restrict__ b_a,
    const __bf16* __restrict__ WfcT, const __bf16* __restrict__ WvT,
    const __bf16* __restrict__ WuT,
    float* __restrict__ hf_out, float* __restrict__ araw_out) {
  __shared__ __bf16 Atile[2][M_ROWS][136];     // double-buffered 64x128 stage
  __shared__ __bf16 feats_s[M_ROWS][H1_ + 8];  // bf16 feats tile for phase B
  __shared__ float  rowsum[M_ROWS];

  const int tid    = threadIdx.x;
  const int lane   = tid & 31;
  const int wave   = tid >> 5;
  const int row0   = blockIdx.x * M_ROWS;
  const int mrow   = lane & 15;       // A row-in-subtile / B,C column index
  const int halfhi = lane >> 4;       // 0: lanes 0-15, 1: lanes 16-31

  // staging geometry: each thread owns 32 consecutive f32 of one h row
  const int sbase = tid * 32;
  const int sr = sbase >> 7;          // staged row 0..63
  const int sc = sbase & 127;         // staged col {0,32,64,96}
  int rg = row0 + sr;
  if (rg > N_PATCH - 1) rg = N_PATCH - 1;  // clamp tail workgroup
  const float* __restrict__ hrow = h + (size_t)rg * D_IN + sc;

  // ---------------- phase A: feats = relu(h @ W_fc + b_fc) ----------------
  v8f acc[MT][4] = {};  // [M subtile][N tile]; 8 waves x 4 N-tiles -> H1=512
  float4 preg[8];       // global->register prefetch of next stage

  // prologue: stage 0 into buffer 0
#pragma unroll
  for (int q = 0; q < 8; ++q)
    preg[q] = *reinterpret_cast<const float4*>(hrow + q * 4);
#pragma unroll
  for (int q = 0; q < 4; ++q) {
    const float4 f0 = preg[2 * q], f1 = preg[2 * q + 1];
    __bf16* dp = &Atile[0][sr][sc + q * 8];
    dp[0] = (__bf16)f0.x; dp[1] = (__bf16)f0.y;
    dp[2] = (__bf16)f0.z; dp[3] = (__bf16)f0.w;
    dp[4] = (__bf16)f1.x; dp[5] = (__bf16)f1.y;
    dp[6] = (__bf16)f1.z; dp[7] = (__bf16)f1.w;
  }

  for (int s = 0; s < NSTAGE; ++s) {
    __syncthreads();  // cur buffer stores visible; prev-stage reads retired
    const int cur = s & 1;
    const int k0 = s * 128;
    if (s + 1 < NSTAGE) {  // issue next-stage global loads before the WMMAs
      const float* gp = hrow + (s + 1) * 128;
      __builtin_prefetch(gp + 128, 0, 1);
#pragma unroll
      for (int q = 0; q < 8; ++q)
        preg[q] = *reinterpret_cast<const float4*>(gp + q * 4);
    }
#pragma unroll
    for (int kk = 0; kk < 4; ++kk) {
      // A frags: lanes<16 hold K[0..7],[16..23]; lanes>=16 hold +8
      const int kbase = kk * 32 + halfhi * 8;
      Frag fa[MT];
#pragma unroll
      for (int mm = 0; mm < MT; ++mm) {
        const __bf16* ap = &Atile[cur][mm * 16 + mrow][kbase];
        fa[mm].q[0] = *reinterpret_cast<const uint4*>(ap);
        fa[mm].q[1] = *reinterpret_cast<const uint4*>(ap + 16);
      }
      const int kb = k0 + kk * 32 + halfhi * 8;
#pragma unroll
      for (int t = 0; t < 4; ++t) {
        const int n0 = (wave * 4 + t) * 16;
        Frag fb;  // B 32x16 bf16: lane holds col n, K[0..7],[16..23] (+8 for lanes>=16)
        const __bf16* bp = WfcT + (size_t)(n0 + mrow) * D_IN + kb;
        fb.q[0] = reinterpret_cast<const uint4*>(bp)[0];
        fb.q[1] = reinterpret_cast<const uint4*>(bp)[2];
#pragma unroll
        for (int mm = 0; mm < MT; ++mm)
          acc[mm][t] = __builtin_amdgcn_wmma_f32_16x16x32_bf16(
              false, fa[mm].v, false, fb.v, (short)0, acc[mm][t], false, false);
      }
    }
    if (s + 1 < NSTAGE) {  // convert + store next stage into alternate buffer
#pragma unroll
      for (int q = 0; q < 4; ++q) {
        const float4 f0 = preg[2 * q], f1 = preg[2 * q + 1];
        __bf16* dp = &Atile[cur ^ 1][sr][sc + q * 8];
        dp[0] = (__bf16)f0.x; dp[1] = (__bf16)f0.y;
        dp[2] = (__bf16)f0.z; dp[3] = (__bf16)f0.w;
        dp[4] = (__bf16)f1.x; dp[5] = (__bf16)f1.y;
        dp[6] = (__bf16)f1.z; dp[7] = (__bf16)f1.w;
      }
    }
  }

  // epilogue A: bias + relu, write fp32 hf to d_out, bf16 copy to LDS
  __syncthreads();  // all WMMA reads of Atile retired before feats_s reuse
#pragma unroll
  for (int t = 0; t < 4; ++t) {
    const int n = (wave * 4 + t) * 16 + mrow;
    const float bias = b_fc[n];
#pragma unroll
    for (int mm = 0; mm < MT; ++mm) {
#pragma unroll
      for (int j = 0; j < 8; ++j) {
        const int m = mm * 16 + j + halfhi * 8;
        float v = acc[mm][t][j] + bias;
        v = v > 0.f ? v : 0.f;
        const int row = row0 + m;
        if (row < N_PATCH) hf_out[(size_t)row * H1_ + n] = v;
        feats_s[m][n] = (__bf16)v;
      }
    }
  }
  if (tid < M_ROWS) rowsum[tid] = 0.f;
  __syncthreads();

  // ---------------- phase B: gated attention scores -----------------------
  v8f cv[MT][2] = {}, cu[MT][2] = {};  // 2 H2 column tiles per wave
  for (int ks = 0; ks < H1_ / 32; ++ks) {
    const int kbase = ks * 32 + halfhi * 8;
    Frag fa[MT];
#pragma unroll
    for (int mm = 0; mm < MT; ++mm) {
      const __bf16* ap = &feats_s[mm * 16 + mrow][kbase];
      fa[mm].q[0] = *reinterpret_cast<const uint4*>(ap);
      fa[mm].q[1] = *reinterpret_cast<const uint4*>(ap + 16);
    }
    const int kb = ks * 32 + halfhi * 8;
#pragma unroll
    for (int t = 0; t < 2; ++t) {
      const int n0 = wave * 32 + t * 16;
      Frag fbv, fbu;
      const __bf16* bpv = WvT + (size_t)(n0 + mrow) * H1_ + kb;
      const __bf16* bpu = WuT + (size_t)(n0 + mrow) * H1_ + kb;
      fbv.q[0] = reinterpret_cast<const uint4*>(bpv)[0];
      fbv.q[1] = reinterpret_cast<const uint4*>(bpv)[2];
#pragma unroll
      for (int mm = 0; mm < MT; ++mm)
        cv[mm][t] = __builtin_amdgcn_wmma_f32_16x16x32_bf16(
            false, fa[mm].v, false, fbv.v, (short)0, cv[mm][t], false, false);
      fbu.q[0] = reinterpret_cast<const uint4*>(bpu)[0];
      fbu.q[1] = reinterpret_cast<const uint4*>(bpu)[2];
#pragma unroll
      for (int mm = 0; mm < MT; ++mm)
        cu[mm][t] = __builtin_amdgcn_wmma_f32_16x16x32_bf16(
            false, fa[mm].v, false, fbu.v, (short)0, cu[mm][t], false, false);
    }
  }
#pragma unroll
  for (int t = 0; t < 2; ++t) {
    const int n = wave * 32 + t * 16 + mrow;
    const float bv = b_v[n], bu = b_u[n], wa = w_a[n];
#pragma unroll
    for (int mm = 0; mm < MT; ++mm) {
#pragma unroll
      for (int j = 0; j < 8; ++j) {
        const int m = mm * 16 + j + halfhi * 8;
        float vv = tanhf(cv[mm][t][j] + bv);
        float uu = 1.0f / (1.0f + __expf(-(cu[mm][t][j] + bu)));
        atomicAdd(&rowsum[m], vv * uu * wa);  // ds_add_f32
      }
    }
  }
  __syncthreads();
  if (tid < M_ROWS && row0 + tid < N_PATCH)
    araw_out[row0 + tid] = rowsum[tid] + b_a[0];
}

// ---------------- kernel 3: softmax stats (max, sum-exp) + zero M ---------
__global__ __launch_bounds__(1024) void softmax_stats(
    const float* __restrict__ araw, float* __restrict__ stats,
    float* __restrict__ Mvec) {
  __shared__ float red[1024];
  const int tid = threadIdx.x;
  float m = -1e30f;
  for (int i = tid; i < N_PATCH; i += 1024) m = fmaxf(m, araw[i]);
  red[tid] = m; __syncthreads();
  for (int s = 512; s > 0; s >>= 1) {
    if (tid < s) red[tid] = fmaxf(red[tid], red[tid + s]);
    __syncthreads();
  }
  const float amax = red[0];
  __syncthreads();
  float z = 0.f;
  for (int i = tid; i < N_PATCH; i += 1024) z += __expf(araw[i] - amax);
  red[tid] = z; __syncthreads();
  for (int s = 512; s > 0; s >>= 1) {
    if (tid < s) red[tid] += red[tid + s];
    __syncthreads();
  }
  if (tid == 0) { stats[0] = amax; stats[1] = red[0]; }
  if (tid < 512) Mvec[tid] = 0.f;
}

// ---------------- kernel 4: M = softmax(A_raw)^T @ hf ---------------------
__global__ __launch_bounds__(256) void bag_pool(
    const float* __restrict__ araw, const float* __restrict__ hf,
    const float* __restrict__ stats, float* __restrict__ Mvec) {
  const int rows_per = (N_PATCH + gridDim.x - 1) / gridDim.x;
  const int start = blockIdx.x * rows_per;
  const int end = (start + rows_per < N_PATCH) ? start + rows_per : N_PATCH;
  const float amax = stats[0], invZ = 1.0f / stats[1];
  const int f0 = threadIdx.x, f1 = threadIdx.x + 256;
  float a0 = 0.f, a1 = 0.f;
  for (int n = start; n < end; ++n) {
    const float a = __expf(araw[n] - amax) * invZ;
    const float* hr = hf + (size_t)n * H1_;
    a0 += a * hr[f0];
    a1 += a * hr[f1];
  }
  unsafeAtomicAdd(&Mvec[f0], a0);  // global_atomic_add_f32
  unsafeAtomicAdd(&Mvec[f1], a1);
}

// ---------------- kernel 5: top-K / bottom-K / mid selection --------------
__global__ __launch_bounds__(1024) void select_topk(
    const float* __restrict__ araw, int* __restrict__ idxbuf,
    unsigned char* __restrict__ flags) {
  __shared__ float rv[1024];
  __shared__ int   ri[1024];
  const int tid = threadIdx.x;
  for (int i = tid; i < N_PATCH; i += 1024) flags[i] = 0;
  __threadfence();
  __syncthreads();
  for (int pass = 0; pass < 2; ++pass) {        // 0: top-K, 1: bottom-K
    for (int k = 0; k < K_SEL; ++k) {
      float bvv = pass ? 1e30f : -1e30f;
      int bi = N_PATCH;
      for (int i = tid; i < N_PATCH; i += 1024) {
        if (flags[i]) continue;
        const float v = araw[i];
        const bool better = pass ? (v < bvv || (v == bvv && i < bi))
                                 : (v > bvv || (v == bvv && i < bi));
        if (better) { bvv = v; bi = i; }
      }
      rv[tid] = bvv; ri[tid] = bi; __syncthreads();
      for (int s = 512; s > 0; s >>= 1) {
        if (tid < s) {
          const bool take = pass
              ? (rv[tid + s] < rv[tid] || (rv[tid + s] == rv[tid] && ri[tid + s] < ri[tid]))
              : (rv[tid + s] > rv[tid] || (rv[tid + s] == rv[tid] && ri[tid + s] < ri[tid]));
          if (take) { rv[tid] = rv[tid + s]; ri[tid] = ri[tid + s]; }
        }
        __syncthreads();
      }
      if (tid == 0) {
        idxbuf[pass * K_SEL + k] = ri[0];
        flags[ri[0]] = 1;
        __threadfence();
      }
      __syncthreads();
    }
  }
  if (tid == 0) {  // mid: K pseudo-random non-excluded indices (LCG)
    unsigned s42 = 42u;
    int cnt = 0;
    while (cnt < K_SEL) {
      s42 = s42 * 1664525u + 1013904223u;
      const int i = (int)(s42 % N_PATCH);
      if (!flags[i]) { flags[i] = 1; idxbuf[2 * K_SEL + cnt] = i; ++cnt; }
    }
  }
}

// ---------------- kernel 6: instance + bag heads --------------------------
__global__ __launch_bounds__(64) void heads(
    const float* __restrict__ hf, const float* __restrict__ Mvec,
    const int* __restrict__ idxbuf,
    const float* __restrict__ W_cls, const float* __restrict__ b_cls,
    const float* __restrict__ W_inst0, const float* __restrict__ b_inst0,
    const float* __restrict__ W_inst1, const float* __restrict__ b_inst1,
    float* __restrict__ out) {
  __shared__ float lsum[64];
  const int tid = threadIdx.x;
  float myloss = 0.f;
  if (tid < 3 * K_SEL) {
    const int idx = idxbuf[tid];
    const float* hr = hf + (size_t)idx * H1_;
    float l0 = b_inst1[0], l1 = b_inst1[1];
    for (int f = 0; f < H1_; ++f) {
      const float x = hr[f];
      l0 += x * W_inst1[f * 2 + 0];
      l1 += x * W_inst1[f * 2 + 1];
    }
    const float mx = fmaxf(l0, l1);
    const float e0 = __expf(l0 - mx), e1 = __expf(l1 - mx);
    const float p0 = e0 / (e0 + e1), p1 = e1 / (e0 + e1);
    const float t1 = (tid < K_SEL) ? 1.f : 0.f;  // top -> class 1
    const float t0 = 1.f - t1;
    myloss = (p0 - t0) * (p0 - t0) + (p1 - t1) * (p1 - t1);
    out[OUT_PREDS1 + tid] = (l1 > l0) ? 1.f : 0.f;
  }
  lsum[tid] = myloss; __syncthreads();
  for (int s = 32; s > 0; s >>= 1) {
    if (tid < s) lsum[tid] += lsum[tid + s];
    __syncthreads();
  }
  if (tid == 0) {
    const float loss1 = lsum[0] / (float)(3 * K_SEL * 2);
    // out-of-class head on the K-th largest attention instance
    const int idx = idxbuf[K_SEL - 1];
    const float* hr = hf + (size_t)idx * H1_;
    float l0 = b_inst0[0], l1 = b_inst0[1];
    for (int f = 0; f < H1_; ++f) {
      const float x = hr[f];
      l0 += x * W_inst0[f * 2 + 0];
      l1 += x * W_inst0[f * 2 + 1];
    }
    const float mx = fmaxf(l0, l1);
    const float e0 = __expf(l0 - mx), e1 = __expf(l1 - mx);
    const float p0 = e0 / (e0 + e1), p1 = e1 / (e0 + e1);
    const float loss0 = ((p0 - 1.f) * (p0 - 1.f) + p1 * p1) * 0.5f;
    out[OUT_ILOSS]  = loss0 + loss1;
    out[OUT_PREDS0] = (l1 > l0) ? 1.f : 0.f;
    // bag head: logits = M @ W_cls + b_cls
    float g0 = b_cls[0], g1 = b_cls[1];
    for (int f = 0; f < H1_; ++f) {
      const float m = Mvec[f];
      g0 += m * W_cls[f * 2 + 0];
      g1 += m * W_cls[f * 2 + 1];
    }
    out[OUT_LOGITS + 0] = g0;
    out[OUT_LOGITS + 1] = g1;
    const float mx2 = fmaxf(g0, g1);
    const float q0 = __expf(g0 - mx2), q1 = __expf(g1 - mx2);
    out[OUT_YPROB + 0] = q0 / (q0 + q1);
    out[OUT_YPROB + 1] = q1 / (q0 + q1);
    out[OUT_YHAT] = (g1 > g0) ? 1.f : 0.f;
  }
}

// ---------------- launcher ----------------
extern "C" void kernel_launch(void* const* d_in, const int* in_sizes, int n_in,
                              void* d_out, int out_size, void* d_ws, size_t ws_size,
                              hipStream_t stream) {
  const float* h       = (const float*)d_in[0];
  const float* W_fc    = (const float*)d_in[1];
  const float* b_fc    = (const float*)d_in[2];
  const float* W_v     = (const float*)d_in[3];
  const float* b_v     = (const float*)d_in[4];
  const float* W_u     = (const float*)d_in[5];
  const float* b_u     = (const float*)d_in[6];
  const float* w_a     = (const float*)d_in[7];
  const float* b_a     = (const float*)d_in[8];
  const float* W_cls   = (const float*)d_in[9];
  const float* b_cls   = (const float*)d_in[10];
  const float* W_inst0 = (const float*)d_in[11];
  const float* b_inst0 = (const float*)d_in[12];
  const float* W_inst1 = (const float*)d_in[13];
  const float* b_inst1 = (const float*)d_in[14];

  char* ws = (char*)d_ws;
  __bf16* WfcT   = (__bf16*)(ws + WS_WFCT);
  __bf16* WvT    = (__bf16*)(ws + WS_WVT);
  __bf16* WuT    = (__bf16*)(ws + WS_WUT);
  float*  stats  = (float*)(ws + WS_STATS);
  float*  Mvec   = (float*)(ws + WS_M);
  int*    idxbuf = (int*)(ws + WS_IDX);
  unsigned char* flags = (unsigned char*)(ws + WS_FLAG);

  float* out      = (float*)d_out;
  float* hf_out   = out + OUT_HF;
  float* araw_out = out + OUT_ARAW;

  transpose_cvt<<<(D_IN * H1_ + 255) / 256, 256, 0, stream>>>(W_fc, WfcT, D_IN, H1_);
  transpose_cvt<<<(H1_ * H2_ + 255) / 256, 256, 0, stream>>>(W_v, WvT, H1_, H2_);
  transpose_cvt<<<(H1_ * H2_ + 255) / 256, 256, 0, stream>>>(W_u, WuT, H1_, H2_);

  fused_attn<<<(N_PATCH + M_ROWS - 1) / M_ROWS, 256, 0, stream>>>(
      h, b_fc, b_v, b_u, w_a, b_a, WfcT, WvT, WuT, hf_out, araw_out);

  softmax_stats<<<1, 1024, 0, stream>>>(araw_out, stats, Mvec);
  bag_pool<<<256, 256, 0, stream>>>(araw_out, hf_out, stats, Mvec);
  select_topk<<<1, 1024, 0, stream>>>(araw_out, idxbuf, flags);
  heads<<<1, 64, 0, stream>>>(hf_out, Mvec, idxbuf, W_cls, b_cls,
                              W_inst0, b_inst0, W_inst1, b_inst1, out);
}
